// HybridGGNNBiLSTM_56221121904781
// MI455X (gfx1250) — matrix operand order repaired and checked by reading, not compiled, not run on hardware
//
#include <hip/hip_runtime.h>
#include <hip/hip_bf16.h>
#include <stdint.h>

typedef __attribute__((ext_vector_type(16))) _Float16 v16h;
typedef __attribute__((ext_vector_type(8)))  float    v8f;

#define BB 32
#define NN 1024
#define FF 34
#define HH 64
#define TT 1024

__device__ __forceinline__ float sigmf(float x) { return 1.f / (1.f + expf(-x)); }

__device__ __forceinline__ v8f wmma_f16f32(v16h a, v16h b, v8f c) {
  // D = A(16x32 f16) * B(32x16 f16) + C(16x16 f32)
  return __builtin_amdgcn_wmma_f32_16x16x32_f16(false, a, false, b, (short)0, c,
                                                false, false);
}

// A fragment 16x32 from row-major f32 (ld elems). Layout: lane<16 holds
// row (lane&15), K 0..15; lane>=16 same row, K 16..31.
__device__ __forceinline__ v16h frag_a_f32(const float* p, int ld) {
  int lane = threadIdx.x & 31;
  const float* q = p + (size_t)(lane & 15) * ld + ((lane >> 4) << 4);
  v16h f;
#pragma unroll
  for (int e = 0; e < 16; ++e) f[e] = (_Float16)q[e];
  return f;
}

// B fragment 32x16; source stored transposed (row = output column, ld elems).
// Contiguous-k layout, same as A: lane<16 holds col (lane&15) K 0..15; lane>=16 K 16..31.
__device__ __forceinline__ v16h frag_b_f16t(const _Float16* p, int ld) {
  int lane = threadIdx.x & 31;
  const _Float16* q = p + (size_t)(lane & 15) * ld + ((lane >> 4) << 4);
  v16h f;
#pragma unroll
  for (int e = 0; e < 16; ++e) f[e] = q[e];
  return f;
}

// ---------------- GGNN ----------------

__global__ void k_proj(const float* __restrict__ nf, const float* __restrict__ pw,
                       const float* __restrict__ pb, float* __restrict__ h) {
  int id = blockIdx.x * blockDim.x + threadIdx.x; // B*N*H
  int d = id & 63;
  int row = id >> 6;
  const float* x = nf + (size_t)row * FF;
  float a = pb[d];
#pragma unroll
  for (int f = 0; f < FF; ++f) a += x[f] * pw[f * HH + d];
  h[id] = fmaxf(a, 0.f);
}

// hwT[e][b][d][j] = sum_k h[b][j][k] * ew[e][k][d]  (f16, transposed for WMMA B)
__global__ void k_edge(const float* __restrict__ h, const float* __restrict__ ew,
                       _Float16* __restrict__ hwT) {
  int id = blockIdx.x * blockDim.x + threadIdx.x; // 2*B*N*H
  int d = id & 63;
  int j = (id >> 6) & 1023;
  int b = (id >> 16) & 31;
  int e = id >> 21;
  const float* hr = h + ((size_t)(b << 10) + j) * HH;
  const float* w = ew + (size_t)e * HH * HH + d;
  float a = 0.f;
#pragma unroll 8
  for (int k = 0; k < HH; ++k) a += hr[k] * w[(size_t)k * HH];
  hwT[((size_t)(e * BB + b) * HH + d) * NN + j] = (_Float16)a;
}

// msg[b, i, :] = adjD[b,i,:] @ hw0 + adjC[b,i,:] @ hw1   (WMMA, f16 math, f32 acc)
__global__ void k_bmm(const float* __restrict__ adjD, const float* __restrict__ adjC,
                      const _Float16* __restrict__ hwT, float* __restrict__ msg) {
  int itile = blockIdx.x;        // 0..63
  int b = blockIdx.y;            // 0..31
  int wave = threadIdx.x >> 5;   // 0..3 -> n-tile
  const float* aD = adjD + (size_t)b * NN * NN + (size_t)itile * 16 * NN;
  const float* aC = adjC + (size_t)b * NN * NN + (size_t)itile * 16 * NN;
  const _Float16* b0 = hwT + ((size_t)b * HH + wave * 16) * NN;            // e=0
  const _Float16* b1 = hwT + ((size_t)(BB + b) * HH + wave * 16) * NN;     // e=1
  v8f acc = {};
  for (int k = 0; k < NN; k += 32) {
    __builtin_prefetch(aD + k + 32, 0, 1);
    __builtin_prefetch(aC + k + 32, 0, 1);
    acc = wmma_f16f32(frag_a_f32(aD + k, NN), frag_b_f16t(b0 + k, NN), acc);
    acc = wmma_f16f32(frag_a_f32(aC + k, NN), frag_b_f16t(b1 + k, NN), acc);
  }
  int lane = threadIdx.x & 31;
  int n = lane & 15, mh = (lane >> 4) << 3;
  float* o = msg + ((size_t)b * NN + itile * 16) * HH + wave * 16;
#pragma unroll
  for (int r = 0; r < 8; ++r) o[(size_t)(mh + r) * HH + n] = acc[r];
}

// GRU cell + residual LayerNorm, one 64-thread block per (b, node) row, h in place.
__global__ void k_gru_ln(const float* __restrict__ msg, float* __restrict__ h,
                         const float* __restrict__ wi, const float* __restrict__ wh,
                         const float* __restrict__ bi, const float* __restrict__ bh,
                         const float* __restrict__ lns, const float* __restrict__ lnb) {
  int row = blockIdx.x;
  int d = threadIdx.x;
  __shared__ float sm[64], sh[64], sr[64], sr2[64];
  const float* mrow = msg + (size_t)row * HH;
  float* hrow = h + (size_t)row * HH;
  sm[d] = mrow[d];
  sh[d] = hrow[d];
  __syncthreads();
  float xr = bi[d], xz = bi[64 + d], xn = bi[128 + d];
  float hr = bh[d], hz = bh[64 + d], hn = bh[128 + d];
#pragma unroll 4
  for (int k = 0; k < HH; ++k) {
    float m = sm[k], hv = sh[k];
    const float* wik = wi + k * 192;
    const float* whk = wh + k * 192;
    xr += m * wik[d];  xz += m * wik[64 + d];  xn += m * wik[128 + d];
    hr += hv * whk[d]; hz += hv * whk[64 + d]; hn += hv * whk[128 + d];
  }
  float r = sigmf(xr + hr), z = sigmf(xz + hz), n = tanhf(xn + r * hn);
  float hnew = (1.f - z) * n + z * sh[d];
  float v = sh[d] + hnew;
  sr[d] = v; sr2[d] = v * v;
  __syncthreads();
  for (int s = 32; s > 0; s >>= 1) {
    if (d < s) { sr[d] += sr[d + s]; sr2[d] += sr2[d + s]; }
    __syncthreads();
  }
  float mu = sr[0] * (1.f / 64.f);
  float var = sr2[0] * (1.f / 64.f) - mu * mu;
  hrow[d] = (v - mu) * rsqrtf(var + 1e-5f) * lns[d] + lnb[d];
}

__global__ void k_gather(const float* __restrict__ h, const int* __restrict__ topo,
                         const float* __restrict__ mask, float* __restrict__ x0) {
  int id = blockIdx.x * blockDim.x + threadIdx.x; // B*N*H
  int d = id & 63;
  int t = (id >> 6) & 1023;
  int b = id >> 16;
  int p = topo[(b << 10) + t];
  x0[id] = h[(((size_t)(b << 10) + p) << 6) + d] * mask[(b << 10) + p];
}

// ---------------- BiLSTM ----------------

__global__ void k_transpose_w(const float* __restrict__ wih, _Float16* __restrict__ wT,
                              int K) {
  int id = blockIdx.x * blockDim.x + threadIdx.x;
  if (id >= 256 * K) return;
  int c = id / K, k = id - c * K;
  wT[(size_t)c * K + k] = (_Float16)wih[(size_t)k * 256 + c];
}

// xg[row, 0:256] = x[row, 0:K] @ w_ih + bias   (WMMA; 8 waves x 2 n-tiles)
__global__ void k_gates(const float* __restrict__ x, const _Float16* __restrict__ wT,
                        const float* __restrict__ bias, float* __restrict__ xg, int K) {
  int mt = blockIdx.x;           // 0..2047 (16-row tiles of 32768 rows)
  int wave = threadIdx.x >> 5;   // 0..7
  const float* A = x + (size_t)mt * 16 * K;
  const _Float16* B0 = wT + (size_t)(wave * 2 + 0) * 16 * K;
  const _Float16* B1 = wT + (size_t)(wave * 2 + 1) * 16 * K;
  v8f acc0 = {}; v8f acc1 = {};
  for (int k = 0; k < K; k += 32) {
    v16h fa = frag_a_f32(A + k, K);
    acc0 = wmma_f16f32(fa, frag_b_f16t(B0 + k, K), acc0);
    acc1 = wmma_f16f32(fa, frag_b_f16t(B1 + k, K), acc1);
  }
  int lane = threadIdx.x & 31;
  int n = lane & 15, mh = (lane >> 4) << 3;
  int n0 = (wave * 2) * 16, n1 = n0 + 16;
  float* orow = xg + (size_t)mt * 16 * 256;
#pragma unroll
  for (int r = 0; r < 8; ++r) {
    orow[(size_t)(mh + r) * 256 + n0 + n] = acc0[r] + bias[n0 + n];
    orow[(size_t)(mh + r) * 256 + n1 + n] = acc1[r] + bias[n1 + n];
  }
}

// Persistent scan: 2 blocks x 1024 threads (16 batches x 64 hidden each).
// W_hh staged transposed in LDS (f16, 32KB). Gate order i,f,g,o.
__global__ void __launch_bounds__(1024) k_lstm_scan(
    const float* __restrict__ xg, const float* __restrict__ whh,
    float* __restrict__ xout, float* __restrict__ finals, int coff, int reverse) {
  __shared__ _Float16 s_w[256 * 64];  // [col][k]
  __shared__ float s_h[16 * 64];
  int tid = threadIdx.x;
  int bl = tid >> 6, j = tid & 63;
  int b = blockIdx.x * 16 + bl;
  for (int i = tid; i < 256 * 64; i += 1024) {
    int c = i >> 6, k = i & 63;
    s_w[i] = (_Float16)whh[(size_t)k * 256 + c];
  }
  s_h[tid] = 0.f;
  __syncthreads();
  float cst = 0.f, hv = 0.f;
  const _Float16* wi = s_w + (size_t)(0 + j) * 64;
  const _Float16* wf = s_w + (size_t)(64 + j) * 64;
  const _Float16* wg = s_w + (size_t)(128 + j) * 64;
  const _Float16* wo = s_w + (size_t)(192 + j) * 64;
  const float* hp = s_h + bl * 64;
  for (int s = 0; s < TT; ++s) {
    int t = reverse ? (TT - 1 - s) : s;
    const float* g = xg + ((size_t)b * TT + t) * 256;
    float gi = g[j], gf = g[64 + j], gg = g[128 + j], go = g[192 + j];
#pragma unroll 8
    for (int k = 0; k < 64; ++k) {
      float hk = hp[k];
      gi += hk * (float)wi[k];
      gf += hk * (float)wf[k];
      gg += hk * (float)wg[k];
      go += hk * (float)wo[k];
    }
    cst = sigmf(gf) * cst + sigmf(gi) * tanhf(gg);
    hv = sigmf(go) * tanhf(cst);
    xout[((size_t)b * TT + t) * 128 + coff + j] = hv;
    __syncthreads();
    s_h[bl * 64 + j] = hv;
    __syncthreads();
  }
  finals[b * 128 + coff + j] = hv;
}

// ---------------- Attention pooling + classifier ----------------

__global__ void k_attn(const float* __restrict__ h_all, const float* __restrict__ w1,
                       const float* __restrict__ b1, const float* __restrict__ w2,
                       const float* __restrict__ b2, float* __restrict__ scores) {
  int id = blockIdx.x * blockDim.x + threadIdx.x; // B*T
  const float* row = h_all + (size_t)id * 128;
  float sc = b2[0];
  for (int u = 0; u < 64; ++u) {
    float a = b1[u];
#pragma unroll 8
    for (int d = 0; d < 128; ++d) a += row[d] * w1[d * 64 + u];
    sc += tanhf(a) * w2[u];
  }
  scores[id] = sc;
}

__global__ void k_pool_mlp(const float* __restrict__ h_all, const float* __restrict__ scores,
                           const int* __restrict__ slen, const float* __restrict__ finals,
                           const float* __restrict__ w1, const float* __restrict__ b1,
                           const float* __restrict__ w2, const float* __restrict__ b2,
                           const float* __restrict__ w3, const float* __restrict__ b3,
                           float* __restrict__ out) {
  int b = blockIdx.x;
  int tid = threadIdx.x; // 256
  __shared__ float red[256];
  __shared__ float wgt[1024];
  __shared__ float hc[128];
  __shared__ float s1[128];
  __shared__ float s2[64];
  int L = slen[b];
  const float* sc = scores + (size_t)b * TT;
  float mx = -1e30f;
  for (int t = tid; t < TT; t += 256)
    if (t < L) mx = fmaxf(mx, sc[t]);
  red[tid] = mx; __syncthreads();
  for (int s = 128; s > 0; s >>= 1) {
    if (tid < s) red[tid] = fmaxf(red[tid], red[tid + s]);
    __syncthreads();
  }
  mx = red[0]; __syncthreads();
  float se = 0.f;
  for (int t = tid; t < TT; t += 256) {
    float w = (t < L) ? expf(sc[t] - mx) : 0.f;
    wgt[t] = w; se += w;
  }
  red[tid] = se; __syncthreads();
  for (int s = 128; s > 0; s >>= 1) {
    if (tid < s) red[tid] += red[tid + s];
    __syncthreads();
  }
  se = red[0];
  float inv = (se > 0.f) ? 1.f / se : 0.f; // nan_to_num guard
  __syncthreads();
  if (tid < 128) {
    float acc = 0.f;
    const float* ha = h_all + (size_t)b * TT * 128 + tid;
    for (int t = 0; t < TT; ++t) acc += wgt[t] * ha[(size_t)t * 128];
    hc[tid] = acc * inv + finals[b * 128 + tid];
  }
  __syncthreads();
  if (tid < 128) {
    float a = b1[tid];
    for (int d = 0; d < 128; ++d) a += hc[d] * w1[d * 128 + tid];
    s1[tid] = fmaxf(a, 0.f);
  }
  __syncthreads();
  if (tid < 64) {
    float a = b2[tid];
    for (int u = 0; u < 128; ++u) a += s1[u] * w2[u * 64 + tid];
    s2[tid] = fmaxf(a, 0.f);
  }
  __syncthreads();
  if (tid < 9) {
    float a = b3[tid];
    for (int v = 0; v < 64; ++v) a += s2[v] * w3[v * 9 + tid];
    out[b * 9 + tid] = a;
  }
}

// ---------------- host ----------------

extern "C" void kernel_launch(void* const* d_in, const int* in_sizes, int n_in,
                              void* d_out, int out_size, void* d_ws, size_t ws_size,
                              hipStream_t stream) {
  (void)in_sizes; (void)n_in; (void)out_size; (void)ws_size;
  // Inputs: recursive insertion-order flatten of setup_inputs().
  const float* nf     = (const float*)d_in[0];
  const float* adjD   = (const float*)d_in[1];
  const float* adjC   = (const float*)d_in[2];
  const int*   topo   = (const int*)  d_in[3];
  const float* mask   = (const float*)d_in[4];
  const int*   slen   = (const int*)  d_in[5];
  const float* proj_w = (const float*)d_in[6];
  const float* proj_b = (const float*)d_in[7];
  const float* edge_w = (const float*)d_in[8];
  const float* gru_wi = (const float*)d_in[9];
  const float* gru_wh = (const float*)d_in[10];
  const float* gru_bi = (const float*)d_in[11];
  const float* gru_bh = (const float*)d_in[12];
  const float* ln_s   = (const float*)d_in[13];
  const float* ln_b   = (const float*)d_in[14];
  const float* l_wih[4] = {(const float*)d_in[15], (const float*)d_in[18],
                           (const float*)d_in[21], (const float*)d_in[24]};
  const float* l_whh[4] = {(const float*)d_in[16], (const float*)d_in[19],
                           (const float*)d_in[22], (const float*)d_in[25]};
  const float* l_b[4]   = {(const float*)d_in[17], (const float*)d_in[20],
                           (const float*)d_in[23], (const float*)d_in[26]};
  const float* attn_w1 = (const float*)d_in[27];
  const float* attn_b1 = (const float*)d_in[28];
  const float* attn_w2 = (const float*)d_in[29];
  const float* attn_b2 = (const float*)d_in[30];
  const float* c_w1 = (const float*)d_in[31];
  const float* c_b1 = (const float*)d_in[32];
  const float* c_w2 = (const float*)d_in[33];
  const float* c_b2 = (const float*)d_in[34];
  const float* c_w3 = (const float*)d_in[35];
  const float* c_b3 = (const float*)d_in[36];

  char* ws = (char*)d_ws;
  float*    h      = (float*)(ws);                              // 8 MB
  float*    msg    = (float*)(ws + ((size_t)8  << 20));         // 8 MB
  _Float16* hwT    = (_Float16*)(ws + ((size_t)16 << 20));      // 8 MB
  float*    x0     = (float*)(ws + ((size_t)24 << 20));         // 8 MB
  float*    x1     = (float*)(ws + ((size_t)32 << 20));         // 16 MB
  float*    x2     = (float*)(ws + ((size_t)48 << 20));         // 16 MB
  float*    xg     = (float*)(ws + ((size_t)64 << 20));         // 32 MB
  float*    scores = (float*)(ws + ((size_t)96 << 20));         // 128 KB
  float*    finals = (float*)(ws + ((size_t)96 << 20) + 0x20000);
  _Float16* wT     = (_Float16*)(ws + ((size_t)96 << 20) + 0x30000);

  const int TOT = BB * NN * HH; // 2M
  k_proj<<<TOT / 256, 256, 0, stream>>>(nf, proj_w, proj_b, h);
  for (int t = 0; t < 4; ++t) {
    k_edge<<<(2 * TOT) / 256, 256, 0, stream>>>(h, edge_w, hwT);
    k_bmm<<<dim3(NN / 16, BB), 128, 0, stream>>>(adjD, adjC, hwT, msg);
    k_gru_ln<<<BB * NN, 64, 0, stream>>>(msg, h, gru_wi, gru_wh, gru_bi, gru_bh,
                                         ln_s + t * HH, ln_b + t * HH);
  }
  k_gather<<<TOT / 256, 256, 0, stream>>>(h, topo, mask, x0);

  const float* xin[2] = {x0, x1};
  float* xout[2] = {x1, x2};
  const int Ks[2] = {64, 128};
  for (int layer = 0; layer < 2; ++layer) {
    for (int dir = 0; dir < 2; ++dir) {
      int idx = layer * 2 + dir;
      int K = Ks[layer];
      k_transpose_w<<<(256 * K + 255) / 256, 256, 0, stream>>>(l_wih[idx], wT, K);
      k_gates<<<BB * NN / 16, 256, 0, stream>>>(xin[layer], wT, l_b[idx], xg, K);
      k_lstm_scan<<<2, 1024, 0, stream>>>(xg, l_whh[idx], xout[layer], finals,
                                          dir * 64, dir);
    }
  }
  k_attn<<<(BB * NN) / 256, 256, 0, stream>>>(x2, attn_w1, attn_b1, attn_w2,
                                              attn_b2, scores);
  k_pool_mlp<<<BB, 256, 0, stream>>>(x2, scores, slen, finals, c_w1, c_b1, c_w2,
                                     c_b2, c_w3, c_b3, (float*)d_out);
}
